// CRF_81123342287663
// MI455X (gfx1250) — compile-verified, matrix operand order checked by reading, not run on hardware
//
#include <hip/hip_runtime.h>
#include <hip/hip_fp16.h>

typedef __attribute__((ext_vector_type(16))) _Float16 v16h;
typedef __attribute__((ext_vector_type(8)))  float    v8f;
typedef __attribute__((ext_vector_type(4)))  unsigned int v4u;
typedef __attribute__((ext_vector_type(8)))  int      v8i;
typedef __attribute__((ext_vector_type(4)))  int      v4i;

#define B_   256
#define T_   1024
#define S_   96
#define STARTIDX 1
#define NEGV -9999.0f
#define NWAVES 6
#define WGSZ (NWAVES * 32)   // 192 threads, 16 rows x 12 col-segments

#define LOG2E 1.4426950408889634f
#define LN2   0.6931471805599453f

// Raw hardware transcendentals: single v_exp_f32 / v_log_f32 + one multiply.
__device__ __forceinline__ float fast_exp(float x) {
    return __builtin_amdgcn_exp2f(x * LOG2E);
}
__device__ __forceinline__ float fast_log(float x) {
    return __builtin_amdgcn_logf(x) * LN2;
}

// ---------------------------------------------------------------------------
// Kernel 1: numerator[b] = sum_t (feat[b,t,s_t] + trans[s_t, s_{t-1}]) * mask
// ---------------------------------------------------------------------------
__global__ __launch_bounds__(256) void crf_numer_kernel(
    const float* __restrict__ feat, const int* __restrict__ states,
    const float* __restrict__ mask, const float* __restrict__ trans,
    float* __restrict__ numer)
{
    const int b = blockIdx.x;
    const int tid = threadIdx.x;
    float acc = 0.0f;
    for (int t = tid; t < T_; t += 256) {
        int s  = states[b * T_ + t];
        int sp = (t == 0) ? STARTIDX : states[b * T_ + t - 1];
        float e  = feat[((size_t)(b * T_ + t)) * S_ + s];
        float tr = trans[s * S_ + sp];
        acc += (e + tr) * mask[b * T_ + t];
    }
    __shared__ float red[256];
    red[tid] = acc;
    __syncthreads();
    #pragma unroll
    for (int off = 128; off > 0; off >>= 1) {
        if (tid < off) red[tid] += red[tid + off];
        __syncthreads();
    }
    if (tid == 0) numer[b] = red[0];
}

// ---------------------------------------------------------------------------
// WMMA fragment helpers (wave32 CDNA5 16-bit layouts)
// ---------------------------------------------------------------------------
__device__ __forceinline__ int frag_k(int kg, int h) {
    return (h < 8) ? (8 * kg + h) : (16 + 8 * kg + (h - 8));
}

__device__ __forceinline__ v16h load_a_frag(const _Float16* __restrict__ expA,
                                            int kk, int lane) {
    const int m  = lane & 15;
    const int kg = lane >> 4;
    v16h a;
    #pragma unroll
    for (int h = 0; h < 16; ++h)
        a[h] = expA[m * S_ + kk + frag_k(kg, h)];
    return a;
}

__device__ __forceinline__ v16h load_b_frag(const _Float16* __restrict__ Bm,
                                            int kk, int n0, int lane) {
    const int n  = n0 + (lane & 15);
    const int kg = lane >> 4;
    v16h b;
    #pragma unroll
    for (int h = 0; h < 16; ++h)
        b[h] = Bm[(kk + frag_k(kg, h)) * S_ + n];
    return b;
}

// ---------------------------------------------------------------------------
// TDM prefetch: DMA a (16 rows x 96 f32) feature tile, row stride T_*S_,
// straight into LDS. 2D descriptor, groups 2/3 zero. Tracked by TENSORcnt.
// ---------------------------------------------------------------------------
__device__ __forceinline__ void tdm_prefetch_feat(const float* gsrc, void* ldsDst) {
    const unsigned long long ga = (unsigned long long)(uintptr_t)gsrc;   // byte addr
    const unsigned int       la = (unsigned int)(uintptr_t)ldsDst;       // LDS offset

    v4u g0;
    g0[0] = 1u;                                            // count=1 (valid), user mode
    g0[1] = la;                                            // lds_addr
    g0[2] = (unsigned int)(ga & 0xffffffffu);              // global_addr[31:0]
    g0[3] = (unsigned int)((ga >> 32) & 0x01ffffffu) | (2u << 30); // [56:32] | type=2

    const unsigned int dim0 = S_;                 // tensor_dim0 = 96 elements
    const unsigned int dim1 = 16u;                // tensor_dim1 = 16 rows
    const unsigned long long str0 = (unsigned long long)T_ * S_;  // elem stride

    v8i g1;
    g1[0] = (int)(2u << 16);                                  // data_size=4B (code 2)
    g1[1] = (int)((dim0 & 0xffffu) << 16);                    // tensor_dim0[15:0]
    g1[2] = (int)(((dim0 >> 16) & 0xffffu) | ((dim1 & 0xffffu) << 16));
    g1[3] = (int)(((dim1 >> 16) & 0xffffu) | ((unsigned)S_ << 16)); // tile_dim0=96
    g1[4] = (int)16;                                          // tile_dim1=16, tile_dim2=0
    g1[5] = (int)(str0 & 0xffffffffu);                        // stride0[31:0]
    g1[6] = (int)((str0 >> 32) & 0xffffu);                    // stride0[47:32]
    g1[7] = 0;

    v4i z = {0, 0, 0, 0};
#if __clang_major__ >= 23
    v8i z8 = {0, 0, 0, 0, 0, 0, 0, 0};
    __builtin_amdgcn_tensor_load_to_lds(g0, g1, z, z, z8, 0);
#else
    __builtin_amdgcn_tensor_load_to_lds(g0, g1, z, z, 0);
#endif
}

// ---------------------------------------------------------------------------
// Kernel 2: forward scan for a 16-batch tile. 6 waves; each wave owns one
// 16x16 output tile. B fragments (exp(transition)^T) live in VGPRs across the
// whole t-loop; features arrive by TDM double-buffering; per-step inner loop
// is 6x ds_load_b128 (A frags) + 3x v_wmma_f32_16x16x32_f16.
// ---------------------------------------------------------------------------
__global__ __launch_bounds__(WGSZ, 1) void crf_scan_kernel(
    const float* __restrict__ feat, const float* __restrict__ mask,
    const float* __restrict__ trans, const float* __restrict__ numer,
    float* __restrict__ out)
{
    __shared__ float    alpha[16][S_];                  // 6 KB
    __shared__ _Float16 expA[16][S_];                   // 3 KB
    __shared__ _Float16 Bm[S_][S_];                     // 18 KB  Bm[k][n]=exp(trans[n][k])
    __shared__ __align__(16) float featL[2][16][S_];    // 12 KB TDM double buffer
    __shared__ float maskL[2][16];
    __shared__ float rowmax[16];
    __shared__ float pmax[16][12];

    const int tid  = threadIdx.x;
    const int lane = tid & 31;
    const int wv   = tid >> 5;          // 0..5
    const int n0   = wv * 16;           // this wave's N tile
    const int b0   = blockIdx.x * 16;   // batch tile base

    const int row = tid / 12;           // 0..15
    const int seg = tid % 12;           // 0..11 (8 columns each)

    // ---- init expT (B layout) and alpha0 ----------------------------------
    for (int i = tid; i < S_ * S_; i += WGSZ) {
        int k = i / S_, n = i % S_;
        Bm[k][n] = (_Float16)fast_exp(trans[n * S_ + k]);
    }
    for (int i = tid; i < 16 * S_; i += WGSZ) {
        int m = i / S_, j = i % S_;
        alpha[m][j] = (j == STARTIDX) ? 0.0f : NEGV;
    }

    // ---- TDM prefetch of t=0 tile + mask ----------------------------------
    if (wv == 0) {
        tdm_prefetch_feat(feat + (size_t)b0 * T_ * S_, &featL[0][0][0]);
        __builtin_amdgcn_s_wait_tensorcnt(0);
    }
    if (tid < 16) maskL[0][tid] = mask[(size_t)(b0 + tid) * T_];
    __syncthreads();

    // ---- hoist constant B fragments into registers (held across t-loop) ---
    v16h bf0 = load_b_frag(&Bm[0][0],  0, n0, lane);
    v16h bf1 = load_b_frag(&Bm[0][0], 32, n0, lane);
    v16h bf2 = load_b_frag(&Bm[0][0], 64, n0, lane);

    int buf = 0;
    for (int t = 0; t < T_; ++t) {
        const int nxt = buf ^ 1;
        // kick off next step's TDM transfer (stays in flight during compute)
        if (t + 1 < T_) {
            if (wv == 0)
                tdm_prefetch_feat(feat + ((size_t)b0 * T_ + (t + 1)) * S_,
                                  &featL[nxt][0][0]);
            if (tid < 16)
                maskL[nxt][tid] = mask[(size_t)(b0 + tid) * T_ + (t + 1)];
        }

        // ---- row max of alpha --------------------------------------------
        {
            float mx = -__builtin_inff();
            #pragma unroll
            for (int u = 0; u < 8; ++u)
                mx = fmaxf(mx, alpha[row][seg * 8 + u]);
            pmax[row][seg] = mx;
        }
        __syncthreads();
        if (tid < 16) {
            float mm = pmax[tid][0];
            #pragma unroll
            for (int k = 1; k < 12; ++k) mm = fmaxf(mm, pmax[tid][k]);
            rowmax[tid] = mm;
        }
        __syncthreads();

        // ---- expA = exp(alpha - rowmax) in f16 (raw v_exp_f32) ------------
        {
            const float rm = rowmax[row];
            #pragma unroll
            for (int u = 0; u < 8; ++u) {
                int j = seg * 8 + u;
                expA[row][j] = (_Float16)fast_exp(alpha[row][j] - rm);
            }
        }
        // wave 0: current buffer's TDM op (issued last iteration) must be done;
        // wait tensorcnt<=1 keeps the freshly issued op for t+1 in flight.
        if (wv == 0) __builtin_amdgcn_s_wait_tensorcnt(1);
        __syncthreads();

        // ---- GEMM tile: C(16x16) = expA(16x96) x Bm(96x16 slice) ----------
        v8f c = {};
        {
            v16h a0 = load_a_frag(&expA[0][0],  0, lane);
            c = __builtin_amdgcn_wmma_f32_16x16x32_f16(false, a0, false, bf0,
                                                       (short)0, c, false, false);
            v16h a1 = load_a_frag(&expA[0][0], 32, lane);
            c = __builtin_amdgcn_wmma_f32_16x16x32_f16(false, a1, false, bf1,
                                                       (short)0, c, false, false);
            v16h a2 = load_a_frag(&expA[0][0], 64, lane);
            c = __builtin_amdgcn_wmma_f32_16x16x32_f16(false, a2, false, bf2,
                                                       (short)0, c, false, false);
        }

        // ---- alpha update: log(C) + rowmax + feat, masked (raw v_log_f32) -
        {
            const int n  = n0 + (lane & 15);
            const int mg = (lane >> 4) * 8;
            #pragma unroll
            for (int r = 0; r < 8; ++r) {
                const int m = mg + r;
                float nv  = fast_log(c[r]) + rowmax[m] + featL[buf][m][n];
                float old = alpha[m][n];
                alpha[m][n] = (maskL[buf][m] > 0.0f) ? nv : old;
            }
        }
        __syncthreads();
        buf = nxt;
    }

    // ---- final logsumexp over states + fuse numerator ---------------------
    if (tid < 16) {
        const int m = tid;
        float mx = -__builtin_inff();
        for (int j = 0; j < S_; ++j) mx = fmaxf(mx, alpha[m][j]);
        float sum = 0.0f;
        for (int j = 0; j < S_; ++j) sum += fast_exp(alpha[m][j] - mx);
        float denom = mx + fast_log(sum);
        out[b0 + m] = denom - numer[b0 + m];
    }
}

// ---------------------------------------------------------------------------
extern "C" void kernel_launch(void* const* d_in, const int* in_sizes, int n_in,
                              void* d_out, int out_size, void* d_ws, size_t ws_size,
                              hipStream_t stream) {
    const float* feat   = (const float*)d_in[0];
    const int*   states = (const int*)d_in[1];
    const float* mask   = (const float*)d_in[2];
    const float* trans  = (const float*)d_in[3];
    float* out   = (float*)d_out;
    float* numer = (float*)d_ws;   // 256 floats of scratch

    crf_numer_kernel<<<B_, 256, 0, stream>>>(feat, states, mask, trans, numer);
    crf_scan_kernel<<<B_ / 16, WGSZ, 0, stream>>>(feat, mask, trans, numer, out);
}
